// WindowAttention_15118284882064
// MI455X (gfx1250) — matrix operand order repaired and checked by reading, not compiled, not run on hardware
//
#include <hip/hip_runtime.h>

typedef _Float16 h16;
typedef __attribute__((ext_vector_type(16))) _Float16 v16h;
typedef __attribute__((ext_vector_type(8)))  _Float16 v8h;
typedef __attribute__((ext_vector_type(8)))  float    v8f;
typedef __attribute__((ext_vector_type(4)))  float    v4f;

#define DIM    256
#define HEADS  8
#define HD     32
#define NTOK   64
#define NWIN   1024
#define NBATCH 4096
#define SCALE  0.17677669529663687f  /* 32^-0.5 */
#define RS     264   /* LDS row stride (halves) for 256-wide matrices (+8 pad) */
#define VTS    72    /* Vt row stride (halves), 64 + 8 pad */

/* ---- WMMA fragment loaders (layouts per CDNA5 ISA §7.12.2) ----
 * A 16x32 f16: lane (l&15)=M; halves K = {0..7,16..23} + 8*(l>=16)  -> two b128
 * B 32x16 f16: lane (l&15)=N; halves K = {0..7,16..23} + 8*(l>=16)  -> two b128
 * C/D 16x16 f32: lane (l&15)=N; VGPR r -> M = r + 8*(l>=16)
 */
__device__ __forceinline__ v16h load_a_frag(const h16* row, int hi, int k0) {
  const v8h a0 = *(const v8h*)(row + k0 + 8 * hi);
  const v8h a1 = *(const v8h*)(row + k0 + 8 * hi + 16);
  v16h r;
#pragma unroll
  for (int i = 0; i < 8; ++i) { r[i] = a0[i]; r[i + 8] = a1[i]; }
  return r;
}

__device__ __forceinline__ v16h load_b_frag(const h16* row, int hi, int k0) {
  const v8h b0 = *(const v8h*)(row + k0 + 8 * hi);
  const v8h b1 = *(const v8h*)(row + k0 + 8 * hi + 16);
  v16h r;
#pragma unroll
  for (int i = 0; i < 8; ++i) { r[i] = b0[i]; r[i + 8] = b1[i]; }
  return r;
}

__device__ __forceinline__ v8f wmma_f16(v16h a, v16h b, v8f c) {
  return __builtin_amdgcn_wmma_f32_16x16x32_f16(false, a, false, b, (short)0, c,
                                                false, false);
}

/* ---- Pre-kernel: f32->f16 weights, densify relative-position bias ---- */
__global__ void prep_kernel(const float* __restrict__ qkvw,
                            const float* __restrict__ projw,
                            const float* __restrict__ table,
                            const int* __restrict__ relidx,
                            h16* __restrict__ qkvw16, h16* __restrict__ projw16,
                            float* __restrict__ rpe) {
  const int t = blockIdx.x * 256 + threadIdx.x;
  if (t < 3 * DIM * DIM) qkvw16[t] = (h16)qkvw[t];
  if (t < DIM * DIM) projw16[t] = (h16)projw[t];
  if (t < HEADS * NTOK * NTOK) {
    const int h = t >> 12;           /* t = h*4096 + n*64 + m */
    const int nm = t & 4095;
    rpe[t] = table[relidx[nm] * HEADS + h];
  }
}

/* ---- Fused window attention: 1 block = 1 window, wave w = head w ---- */
__global__ __launch_bounds__(256) void swin_attn(
    const float* __restrict__ x, const float* __restrict__ mask,
    const h16* __restrict__ qkvw, const float* __restrict__ qkvb,
    const h16* __restrict__ projw, const float* __restrict__ projb,
    const float* __restrict__ rpe, float* __restrict__ out) {
  extern __shared__ char smem[];
  h16* Xs   = (h16*)smem;              /* 64 x RS halves, reused as Outs */
  h16* Qs   = (h16*)(smem + 33792);    /* 64 x RS  (pre-scaled by SCALE) */
  h16* Ks   = (h16*)(smem + 67584);    /* 64 x RS                         */
  h16* Vt   = (h16*)(smem + 101376);   /* 256 x VTS (V transposed)        */
  h16* Outs = Xs;

  const int b    = blockIdx.x;
  const int tid  = threadIdx.x;
  const int lane = tid & 31;
  const int wave = tid >> 5;
  const int col  = lane & 15;
  const int hi   = lane >> 4;

  /* ---------- stage x[b] -> f16 LDS ---------- */
  const float* xb = x + (size_t)b * (NTOK * DIM);
#pragma unroll
  for (int i = 0; i < 16; ++i) {
    const int v  = tid + i * 256;          /* float4 index 0..4095 */
    const int n  = v >> 6;
    const int c4 = (v & 63) << 2;
    const v4f xv = *(const v4f*)(xb + n * DIM + c4);
    h16* d = Xs + n * RS + c4;
    d[0] = (h16)xv[0]; d[1] = (h16)xv[1]; d[2] = (h16)xv[2]; d[3] = (h16)xv[3];
  }

  /* warm the L2/L0 path for this wave's weight rows (global_prefetch_b8) */
#pragma unroll
  for (int part = 0; part < 3; ++part)
    __builtin_prefetch(qkvw + (size_t)(part * 256 + wave * HD + col) * DIM, 0, 3);
  __builtin_prefetch(projw + (size_t)(wave * HD + col) * DIM, 0, 3);
  __syncthreads();

  /* ---------- phase 1: qkv GEMM, wave h produces Q_h, K_h, V_h ---------- */
  {
    const int h = wave;
#pragma unroll
    for (int part = 0; part < 3; ++part) {
#pragma unroll
      for (int sub = 0; sub < 2; ++sub) {
        const int oglob = part * 256 + h * HD + sub * 16;
        const h16* wrow = qkvw + (size_t)(oglob + col) * DIM;
        v16h Bf[8];
#pragma unroll
        for (int k = 0; k < 8; ++k) Bf[k] = load_b_frag(wrow, hi, k * 32);
        const float bias = qkvb[oglob + col];
        const int cc = h * HD + sub * 16 + col;  /* column within 256 */
#pragma unroll
        for (int tm = 0; tm < 4; ++tm) {
          const h16* arow = Xs + (tm * 16 + col) * RS;
          v8f acc = {};
#pragma unroll
          for (int k = 0; k < 8; ++k)
            acc = wmma_f16(load_a_frag(arow, hi, k * 32), Bf[k], acc);
          if (part == 0) {
#pragma unroll
            for (int r = 0; r < 8; ++r)
              Qs[(tm * 16 + 8 * hi + r) * RS + cc] =
                  (h16)((acc[r] + bias) * SCALE);
          } else if (part == 1) {
#pragma unroll
            for (int r = 0; r < 8; ++r)
              Ks[(tm * 16 + 8 * hi + r) * RS + cc] = (h16)(acc[r] + bias);
          } else {  /* V: rows M contiguous in D-frag -> single b128 to Vt */
            v8h pk;
#pragma unroll
            for (int r = 0; r < 8; ++r) pk[r] = (h16)(acc[r] + bias);
            *(v8h*)(Vt + cc * VTS + tm * 16 + 8 * hi) = pk;
          }
        }
      }
    }
  }
  __syncthreads();

  /* ---------- phase 2: attention head h = wave (S^T layout trick) ----- */
  {
    const int h = wave;
    const float* maskw = mask + (size_t)(b & (NWIN - 1)) * (NTOK * NTOK);
    const float* rpeh  = rpe + h * (NTOK * NTOK);

    v16h Ka[4]; /* A-operand of S^T = K rows */
#pragma unroll
    for (int mt = 0; mt < 4; ++mt)
      Ka[mt] = load_a_frag(Ks + (mt * 16 + col) * RS + h * HD, hi, 0);

    v16h Bv[2][2]; /* B-operand of O = V^T rows */
#pragma unroll
    for (int dt = 0; dt < 2; ++dt)
#pragma unroll
      for (int kc = 0; kc < 2; ++kc)
        Bv[dt][kc] =
            load_b_frag(Vt + (h * HD + dt * 16 + col) * VTS, hi, kc * 32);

    const v8f zero = {};
#pragma unroll
    for (int nt = 0; nt < 4; ++nt) {
      const v16h Bq = load_b_frag(Qs + (nt * 16 + col) * RS + h * HD, hi, 0);
      v8f St[4];
#pragma unroll
      for (int mt = 0; mt < 4; ++mt) St[mt] = wmma_f16(Ka[mt], Bq, zero);

      /* bias + mask; lane owns column n of S^T -> row n of attention.
       * m = mt*16 + 8*hi + r runs over 8 contiguous floats -> b128 loads. */
      const int n = nt * 16 + col;
      const float* mrow = maskw + n * NTOK;
      const float* rrow = rpeh + n * NTOK;
      float mx = -3.0e38f;
#pragma unroll
      for (int mt = 0; mt < 4; ++mt) {
        const int m0 = mt * 16 + 8 * hi;
        const v4f ma = *(const v4f*)(mrow + m0);
        const v4f mb = *(const v4f*)(mrow + m0 + 4);
        const v4f ra = *(const v4f*)(rrow + m0);
        const v4f rb = *(const v4f*)(rrow + m0 + 4);
#pragma unroll
        for (int r = 0; r < 4; ++r) {
          const float v = St[mt][r] + ma[r] + ra[r];
          St[mt][r] = v;
          mx = fmaxf(mx, v);
        }
#pragma unroll
        for (int r = 0; r < 4; ++r) {
          const float v = St[mt][r + 4] + mb[r] + rb[r];
          St[mt][r + 4] = v;
          mx = fmaxf(mx, v);
        }
      }
      mx = fmaxf(mx, __shfl_xor(mx, 16, 32));
      float sum = 0.f;
#pragma unroll
      for (int mt = 0; mt < 4; ++mt)
#pragma unroll
        for (int r = 0; r < 8; ++r) {
          const float e = __expf(St[mt][r] - mx);
          St[mt][r] = e;
          sum += e;
        }
      sum += __shfl_xor(sum, 16, 32);
      const float inv = 1.0f / sum;

      /* S^T D-frags (mt=2k,2k+1) ARE the A-frag of P for K-chunk k */
      v16h Pa[2];
#pragma unroll
      for (int kc = 0; kc < 2; ++kc) {
        v16h a;
#pragma unroll
        for (int i = 0; i < 8; ++i) {
          a[i]     = (h16)(St[2 * kc][i] * inv);
          a[i + 8] = (h16)(St[2 * kc + 1][i] * inv);
        }
        Pa[kc] = a;
      }
#pragma unroll
      for (int dt = 0; dt < 2; ++dt) {
        v8f o = zero;
        o = wmma_f16(Pa[0], Bv[dt][0], o);
        o = wmma_f16(Pa[1], Bv[dt][1], o);
#pragma unroll
        for (int r = 0; r < 8; ++r)
          Outs[(nt * 16 + 8 * hi + r) * RS + h * HD + dt * 16 + col] =
              (h16)o[r];
      }
    }
  }
  __syncthreads();

  /* ---------- phase 3: proj GEMM + bias, fp32 out ---------- */
  {
    const int tm = wave & 3;
    const int oo = wave >> 2; /* 0/1: interleaved o-tiles */
    const h16* arow = Outs + (tm * 16 + col) * RS;
    v16h Af[8];
#pragma unroll
    for (int k = 0; k < 8; ++k) Af[k] = load_a_frag(arow, hi, k * 32);
    float* ob = out + (size_t)b * (NTOK * DIM);
#pragma unroll
    for (int i = 0; i < 8; ++i) {
      const int o = (oo + 2 * i) * 16;
      const h16* wrow = projw + (size_t)(o + col) * DIM;
      v8f acc = {};
#pragma unroll
      for (int k = 0; k < 8; ++k)
        acc = wmma_f16(Af[k], load_b_frag(wrow, hi, k * 32), acc);
      const float bias = projb[o + col];
#pragma unroll
      for (int r = 0; r < 8; ++r)
        ob[(tm * 16 + 8 * hi + r) * DIM + o + col] = acc[r] + bias;
    }
  }
}

extern "C" void kernel_launch(void* const* d_in, const int* in_sizes, int n_in,
                              void* d_out, int out_size, void* d_ws,
                              size_t ws_size, hipStream_t stream) {
  (void)in_sizes; (void)n_in; (void)out_size; (void)ws_size;
  const float* x      = (const float*)d_in[0];
  const float* mask   = (const float*)d_in[1];
  const float* qkvw   = (const float*)d_in[2];
  const float* qkvb   = (const float*)d_in[3];
  const float* projw  = (const float*)d_in[4];
  const float* projb  = (const float*)d_in[5];
  const float* table  = (const float*)d_in[6];
  const int*   relidx = (const int*)d_in[7];

  char* ws = (char*)d_ws;
  h16*   qkvw16  = (h16*)ws;                /* 196608 h = 384 KB */
  h16*   projw16 = (h16*)(ws + 393216);     /*  65536 h = 128 KB */
  float* rpe     = (float*)(ws + 524288);   /*  32768 f = 128 KB */

  prep_kernel<<<768, 256, 0, stream>>>(qkvw, projw, table, relidx, qkvw16,
                                       projw16, rpe);
  swin_attn<<<NBATCH, 256, 138240, stream>>>(x, mask, qkvw16, qkvb, projw16,
                                             projb, rpe, (float*)d_out);
}
